// CameraAwareMemory_19765439496774
// MI455X (gfx1250) — matrix operand order repaired and checked by reading, not compiled, not run on hardware
//
#include <hip/hip_runtime.h>
#include <hip/hip_bf16.h>

// ---------------------------------------------------------------------------
// Problem constants (match the JAX reference)
// ---------------------------------------------------------------------------
#define BATCH     256
#define DIM       2048
#define NPROXY    16384
#define PPOS      4
#define KSEL      54          // BG_KNN + P
#define BIGF      1e4f
#define INV_TEMP  20.0f       // 1 / 0.05

typedef _Float16 v8h  __attribute__((ext_vector_type(8)));
typedef _Float16 v16h __attribute__((ext_vector_type(16)));
typedef float    v8f  __attribute__((ext_vector_type(8)));

union HV { v16h v; v8h h[2]; };

// ---------------------------------------------------------------------------
// Kernel 1: score = (features @ em^T) * INV_TEMP   via v_wmma_f32_16x16x32_f16
//   grid  = (NPROXY/128, BATCH/128), block = 256 (8 waves of 32)
//   128x128 tile per workgroup; waves tiled 2(M) x 4(N); each wave 64x32
//   = 4x2 fragments of 16x16.  Double-buffered LDS (one barrier/iteration,
//   global loads of chunk k+1 overlap the WMMAs of chunk k).
// ---------------------------------------------------------------------------
#define TM 128
#define TN 128
#define TK 32
#define RS 40   // LDS row stride in halves (80B, 16B-aligned, conflict-free)

__global__ __launch_bounds__(256)
void cam_gemm_wmma(const float* __restrict__ A,   // [BATCH, DIM] features
                   const float* __restrict__ Bm,  // [NPROXY, DIM] em (row-major)
                   float* __restrict__ C)         // [BATCH, NPROXY] score/TEMP
{
    __shared__ _Float16 As[2][TM * RS];
    __shared__ _Float16 Bs[2][TN * RS];

    const int tid  = threadIdx.x;
    const int m0   = blockIdx.y * TM;
    const int n0   = blockIdx.x * TN;
    const int wave = tid >> 5;
    const int lane = tid & 31;
    const int waveM = (wave >> 2) * 64;   // 0 or 64
    const int waveN = (wave & 3) * 32;    // 0,32,64,96
    const int l  = lane & 15;
    const int hh = lane >> 4;             // half-wave select

    // staging: each thread converts 16 consecutive K values of one row
    const int srow  = tid >> 1;           // 0..127
    const int skoff = (tid & 1) * 16;     // 0 or 16

    v8f acc[4][2] = {};

    const float* agp = A  + (size_t)(m0 + srow) * DIM + skoff;
    const float* bgp = Bm + (size_t)(n0 + srow) * DIM + skoff;

    float4 ra[4], rb[4];
    auto gload = [&](int k0) {
        const float4* ga = (const float4*)(agp + k0);
        const float4* gb = (const float4*)(bgp + k0);
        ra[0] = ga[0]; ra[1] = ga[1]; ra[2] = ga[2]; ra[3] = ga[3];
        rb[0] = gb[0]; rb[1] = gb[1]; rb[2] = gb[2]; rb[3] = gb[3];
    };
    auto cvtstore = [&](int p) {
        v8h alo = { (_Float16)ra[0].x, (_Float16)ra[0].y, (_Float16)ra[0].z, (_Float16)ra[0].w,
                    (_Float16)ra[1].x, (_Float16)ra[1].y, (_Float16)ra[1].z, (_Float16)ra[1].w };
        v8h ahi = { (_Float16)ra[2].x, (_Float16)ra[2].y, (_Float16)ra[2].z, (_Float16)ra[2].w,
                    (_Float16)ra[3].x, (_Float16)ra[3].y, (_Float16)ra[3].z, (_Float16)ra[3].w };
        v8h blo = { (_Float16)rb[0].x, (_Float16)rb[0].y, (_Float16)rb[0].z, (_Float16)rb[0].w,
                    (_Float16)rb[1].x, (_Float16)rb[1].y, (_Float16)rb[1].z, (_Float16)rb[1].w };
        v8h bhi = { (_Float16)rb[2].x, (_Float16)rb[2].y, (_Float16)rb[2].z, (_Float16)rb[2].w,
                    (_Float16)rb[3].x, (_Float16)rb[3].y, (_Float16)rb[3].z, (_Float16)rb[3].w };
        *(v8h*)&As[p][srow * RS + skoff]     = alo;
        *(v8h*)&As[p][srow * RS + skoff + 8] = ahi;
        *(v8h*)&Bs[p][srow * RS + skoff]     = blo;
        *(v8h*)&Bs[p][srow * RS + skoff + 8] = bhi;
    };

    // prologue: stage chunk 0 into buffer 0
    gload(0);
    cvtstore(0);
    int p = 0;

    for (int k0 = 0; k0 < DIM; k0 += TK) {
        __syncthreads();                       // buf[p] staged; buf[p^1] free
        const bool more = (k0 + TK < DIM);
        if (more) {
            if (k0 + 2 * TK < DIM) {           // gfx1250 global_prefetch_b8
                __builtin_prefetch(agp + k0 + 2 * TK, 0, 3);
                __builtin_prefetch(bgp + k0 + 2 * TK, 0, 3);
            }
            gload(k0 + TK);                    // VMEM overlaps the WMMAs below
        }

        // ---- fragments from buf[p] ----
        // A 16x32 f16 layout: lanes 0-15 hold K{0..7,16..23}, lanes 16-31 K{8..15,24..31}
        HV afr[4];
#pragma unroll
        for (int m = 0; m < 4; ++m) {
            const int r = waveM + m * 16 + l;
            afr[m].h[0] = *(const v8h*)&As[p][r * RS + hh * 8];
            afr[m].h[1] = *(const v8h*)&As[p][r * RS + 16 + hh * 8];
        }
        // B 32x16 f16 layout: lane n holds col n, lanes 0-15 K{0..7,16..23}, lanes 16-31 K{8..15,24..31}
        HV bfr[2];
#pragma unroll
        for (int n = 0; n < 2; ++n) {
            const int r = waveN + n * 16 + l;
            bfr[n].h[0] = *(const v8h*)&Bs[p][r * RS + hh * 8];
            bfr[n].h[1] = *(const v8h*)&Bs[p][r * RS + 16 + hh * 8];
        }

#pragma unroll
        for (int m = 0; m < 4; ++m)
#pragma unroll
            for (int n = 0; n < 2; ++n)
                acc[m][n] = __builtin_amdgcn_wmma_f32_16x16x32_f16(
                    false, afr[m].v, false, bfr[n].v,
                    (short)0, acc[m][n], false, false);

        if (more) cvtstore(p ^ 1);             // stage next chunk into free buffer
        p ^= 1;
    }

    // ---- epilogue: C/D 16x16 layout: VGPR r -> row base+r+8*hh, col base+l ----
#pragma unroll
    for (int m = 0; m < 4; ++m)
#pragma unroll
        for (int n = 0; n < 2; ++n) {
            const int col = n0 + waveN + n * 16 + l;
#pragma unroll
            for (int r = 0; r < 8; ++r) {
                const int row = m0 + waveM + m * 16 + hh * 8 + r;
                C[(size_t)row * NPROXY + col] = acc[m][n][r] * INV_TEMP;
            }
        }
}

// ---------------------------------------------------------------------------
// Kernel 2: per-row exact top-(BG_KNN+P) via radix select + log-softmax loss
//   one block (256 threads) per row; whole row cached in LDS (64KB of 320KB)
//   Row is pulled global->LDS with gfx1250 async-to-LDS loads (ASYNCcnt).
// ---------------------------------------------------------------------------
__device__ __forceinline__ unsigned fkey(float f) {
    unsigned u = __float_as_uint(f);
    return (u & 0x80000000u) ? ~u : (u | 0x80000000u);   // monotone: bigger f -> bigger key
}
__device__ __forceinline__ float unkey(unsigned u) {
    return __uint_as_float((u & 0x80000000u) ? (u & 0x7FFFFFFFu) : ~u);
}
__device__ __forceinline__ void lse_merge(float& m, float& s, float v, float w) {
    if (v > m) { s = s * __expf(m - v) + w; m = v; }
    else       { s += w * __expf(v - m); }
}

__global__ __launch_bounds__(256)
void cam_rowloss(const float* __restrict__ score,     // [BATCH, NPROXY] (already /TEMP)
                 const int* __restrict__ targets,     // [BATCH]
                 const int* __restrict__ apl,         // [N_IMG]
                 const int* __restrict__ plt,         // [N_CLUSTER, P]
                 float* __restrict__ rowloss)         // [BATCH]
{
    __shared__ float    vals[NPROXY];      // original (unmodified at non-positive slots)
    __shared__ unsigned hist[256];
    __shared__ float    redm[256], reds[256];
    __shared__ float    posval[PPOS];
    __shared__ int      posidx[PPOS];
    __shared__ unsigned bc[2];             // prefix, remaining-k

    const int row = blockIdx.x;
    const int tid = threadIdx.x;
    const float* src = score + (size_t)row * NPROXY;

    // ---- async copy of the whole row into LDS (global_load_async_to_lds_b128)
    {
        const unsigned ldsbase = (unsigned)(uintptr_t)&vals[0];
        for (int j4 = tid; j4 < NPROXY / 4; j4 += 256) {
            const float*  g     = src + j4 * 4;
            const unsigned laddr = ldsbase + (unsigned)j4 * 16u;
            asm volatile("global_load_async_to_lds_b128 %0, %1, off"
                         :: "v"(laddr), "v"(g) : "memory");
        }
        asm volatile("s_wait_asynccnt 0" ::: "memory");
    }
    __syncthreads();

    if (tid == 0) {
        const int py = apl[targets[row]];
#pragma unroll
        for (int p = 0; p < PPOS; ++p) posidx[p] = plt[py * PPOS + p];
    }
    __syncthreads();
    if (tid < PPOS) posval[tid] = vals[posidx[tid]];   // save originals first
    __syncthreads();
    if (tid < PPOS) vals[posidx[tid]] = BIGF;          // detached copy gets BIG
    __syncthreads();

    // ---- radix select: key of the KSEL-th largest (MSB-first, 4x8-bit) ----
    unsigned prefix = 0, kk = KSEL;
#pragma unroll
    for (int pass = 3; pass >= 0; --pass) {
        hist[tid] = 0;
        __syncthreads();
        const unsigned shift = pass * 8;
        const unsigned pmask = (pass == 3) ? 0u : (0xFFFFFFFFu << (shift + 8));
        for (int j = tid; j < NPROXY; j += 256) {
            const unsigned u = fkey(vals[j]);
            if ((u & pmask) == prefix)
                atomicAdd(&hist[(u >> shift) & 255u], 1u);
        }
        __syncthreads();
        if (tid == 0) {
            unsigned cum = 0;
            for (int b = 255; b >= 0; --b) {
                const unsigned c = hist[b];
                if (cum + c >= kk) {
                    bc[0] = prefix | ((unsigned)b << shift);
                    bc[1] = kk - cum;
                    break;
                }
                cum += c;
            }
        }
        __syncthreads();
        prefix = bc[0];
        kk     = bc[1];
        __syncthreads();
    }
    const unsigned u_t       = prefix;   // key of KSEL-th largest (modified values)
    const unsigned n_eq_take = kk;       // ties taken at the threshold
    const float    v_t       = unkey(u_t);

    // ---- online logsumexp over selected originals ----
    float m = -1e30f, s = 0.0f;
    for (int j = tid; j < NPROXY; j += 256) {
        const float v = vals[j];
        if (fkey(v) > u_t && v != BIGF)          // positives handled separately
            lse_merge(m, s, v, 1.0f);
    }
    if (tid == 0) {
        lse_merge(m, s, v_t, (float)n_eq_take);  // threshold ties
#pragma unroll
        for (int p = 0; p < PPOS; ++p) {         // distinct positive originals
            bool dup = false;
#pragma unroll
            for (int q = 0; q < PPOS; ++q)
                if (q < p) dup |= (posidx[q] == posidx[p]);
            if (!dup) lse_merge(m, s, posval[p], 1.0f);
        }
    }
    redm[tid] = m; reds[tid] = s;
    __syncthreads();
    for (int off = 128; off > 0; off >>= 1) {
        if (tid < off) {
            const float m1 = redm[tid],       s1 = reds[tid];
            const float m2 = redm[tid + off], s2 = reds[tid + off];
            const float mm = m1 > m2 ? m1 : m2;
            redm[tid] = mm;
            reds[tid] = s1 * __expf(m1 - mm) + s2 * __expf(m2 - mm);
        }
        __syncthreads();
    }
    if (tid == 0) {
        const float lse = logf(reds[0]) + redm[0];
        const float sp  = posval[0] + posval[1] + posval[2] + posval[3];
        rowloss[row] = lse - (1.0f / PPOS) * sp;
    }
}

// ---------------------------------------------------------------------------
// Kernel 3: deterministic mean over BATCH row losses
// ---------------------------------------------------------------------------
__global__ __launch_bounds__(256)
void cam_reduce(const float* __restrict__ rowloss, float* __restrict__ out)
{
    __shared__ float sm[256];
    const int tid = threadIdx.x;
    sm[tid] = rowloss[tid];
    __syncthreads();
    for (int off = 128; off > 0; off >>= 1) {
        if (tid < off) sm[tid] += sm[tid + off];
        __syncthreads();
    }
    if (tid == 0) out[0] = sm[0] * (1.0f / BATCH);
}

// ---------------------------------------------------------------------------
extern "C" void kernel_launch(void* const* d_in, const int* in_sizes, int n_in,
                              void* d_out, int out_size, void* d_ws, size_t ws_size,
                              hipStream_t stream) {
    const float* features = (const float*)d_in[0];   // [256, 2048] f32
    const float* em       = (const float*)d_in[1];   // [16384, 2048] f32 (normalized)
    const int*   targets  = (const int*)d_in[2];     // [256]
    const int*   apl      = (const int*)d_in[3];     // [32768]
    const int*   plt      = (const int*)d_in[4];     // [4096, 4]

    float* score   = (float*)d_ws;                         // 16 MB
    float* rowloss = score + (size_t)BATCH * NPROXY;       // 1 KB
    float* out     = (float*)d_out;

    dim3 g1(NPROXY / TN, BATCH / TM);
    cam_gemm_wmma<<<g1, 256, 0, stream>>>(features, em, score);
    cam_rowloss<<<BATCH, 256, 0, stream>>>(score, targets, apl, plt, rowloss);
    cam_reduce<<<1, 256, 0, stream>>>(rowloss, out);
}
